// HyenaBaseBlock_42365557407844
// MI455X (gfx1250) — compile-verified, matrix operand order checked by reading, not run on hardware
//
#include <hip/hip_runtime.h>
#include <math.h>

// ============================================================================
// Hyena base block for MI455X (gfx1250, wave32).
//   B=4, L=4096, D=512, P=128, S=2, FFT length 8192.
// GEMMs: bf16x3 split-precision on V_WMMA_F32_16X16X32_BF16, loads grouped
// ahead of the WMMA burst and 64-bit stream pointers hoisted so the scheduler
// can clause + pipeline (round-2 codegen shape).
// FFT conv: per-channel 8192-pt complex FFT in LDS; inputs staged with
// GLOBAL_LOAD_ASYNC_TO_LDS_B128 (ASYNCcnt).
// ============================================================================

typedef __attribute__((ext_vector_type(4)))  __bf16 v4bf;
typedef __attribute__((ext_vector_type(8)))  __bf16 v8bf;
typedef __attribute__((ext_vector_type(16))) __bf16 v16bf;
typedef __attribute__((ext_vector_type(8)))  float  v8f;

#define LN_EPS 1e-5f

__device__ __forceinline__ v16bf cat16(v8bf a, v8bf b) {
    return __builtin_shufflevector(a, b, 0,1,2,3,4,5,6,7,8,9,10,11,12,13,14,15);
}

// Async memory->LDS copy of 16 bytes (per lane); tracked by ASYNCcnt.
__device__ __forceinline__ void async_load_f128(const float* gsrc, float* lds_dst) {
    const unsigned ldsoff = (unsigned)(size_t)lds_dst;  // low 32b = LDS offset
    asm volatile("global_load_async_to_lds_b128 %0, %1, off"
                 :: "v"(ldsoff), "v"(gsrc) : "memory");
}
__device__ __forceinline__ void wait_async0() {
    asm volatile("s_wait_asynccnt 0" ::: "memory");
}

// ---------------------------------------------------------------------------
// Split fp32 -> bf16 hi/lo planes (hi = RNE bf16, lo = bf16(residual)).
// ---------------------------------------------------------------------------
__global__ __launch_bounds__(256)
void split_bf16(const float* __restrict__ in, __bf16* __restrict__ hi,
                __bf16* __restrict__ lo, int n4)
{
    const int i = blockIdx.x * 256 + threadIdx.x;
    if (i >= n4) return;
    const float4 v = ((const float4*)in)[i];
    const __bf16 h0 = (__bf16)v.x, h1 = (__bf16)v.y, h2 = (__bf16)v.z, h3 = (__bf16)v.w;
    v4bf H = {h0, h1, h2, h3};
    v4bf L = {(__bf16)(v.x - (float)h0), (__bf16)(v.y - (float)h1),
              (__bf16)(v.z - (float)h2), (__bf16)(v.w - (float)h3)};
    ((v4bf*)hi)[i] = H;
    ((v4bf*)lo)[i] = L;
}

// ---------------------------------------------------------------------------
// bf16x3 GEMM: C[M,N] = epilogue( A[M,K] @ W[N,K]^T + bias[N] )
// One wave -> 4 adjacent 16x16 N-tiles; 8 waves/block share the A rows (L1).
// Per K-step of 32: ALL loads first (ah, al, bh[4], bl[4]) then 12 WMMAs.
// flags: 1 = SiLU, 2 = *exp(-row*exp(a0)), 4 = += res.
// ---------------------------------------------------------------------------
__global__ __launch_bounds__(256)
void gemm_bf16x3(const __bf16* __restrict__ Ahi, const __bf16* __restrict__ Alo,
                 const __bf16* __restrict__ Whi, const __bf16* __restrict__ Wlo,
                 const float* __restrict__ bias,
                 float* __restrict__ Cf, __bf16* __restrict__ Chi, __bf16* __restrict__ Clo,
                 const float* __restrict__ res, const float* __restrict__ a_scalar,
                 int M, int N, int K, int flags)
{
    const int wid = threadIdx.x >> 5;
    const int l   = threadIdx.x & 31;
    const int tilesNg = N >> 6;
    const int totalJobs = (M >> 4) * tilesNg;
    const int job = blockIdx.x * 8 + wid;
    if (job >= totalJobs) return;                  // wave-uniform
    const int tm  = job / tilesNg;
    const int tng = job - tm * tilesNg;

    // 16-bit 16x32 operand layout: lane l -> row (l&15); two 8-elem K runs at
    // 8*(l>>4) and 16+8*(l>>4).  Identical addressing for A and B (=W^T).
    const int lm = l & 15;
    const int hs = l >> 4;
    const int kofs = hs * 8;

    // precomputed per-stream pointers (bumped by the unrolled loop)
    const __bf16* pAh = Ahi + (size_t)(tm * 16 + lm) * K + kofs;
    const __bf16* pAl = Alo + (size_t)(tm * 16 + lm) * K + kofs;
    const __bf16* pWh[4];
    const __bf16* pWl[4];
    #pragma unroll
    for (int t = 0; t < 4; ++t) {
        const size_t w = (size_t)((tng * 4 + t) * 16 + lm) * K + kofs;
        pWh[t] = Whi + w;
        pWl[t] = Wlo + w;
    }

    const v8f vzero = {0.f,0.f,0.f,0.f,0.f,0.f,0.f,0.f};
    v8f acc[4] = {vzero, vzero, vzero, vzero};

    #pragma unroll 2
    for (int k0 = 0; k0 < K; k0 += 32) {
        if (k0 + 32 < K) {                          // global_prefetch_b8
            __builtin_prefetch((const void*)(pAh + k0 + 32), 0, 1);
            __builtin_prefetch((const void*)(pAl + k0 + 32), 0, 1);
        }
        // ---- all loads up front (distinct SSA values -> one big clause) ----
        const v16bf ah = cat16(*(const v8bf*)(pAh + k0),
                               *(const v8bf*)(pAh + k0 + 16));
        const v16bf al = cat16(*(const v8bf*)(pAl + k0),
                               *(const v8bf*)(pAl + k0 + 16));
        v16bf bh[4], bl[4];
        #pragma unroll
        for (int t = 0; t < 4; ++t) {
            bh[t] = cat16(*(const v8bf*)(pWh[t] + k0),
                          *(const v8bf*)(pWh[t] + k0 + 16));
            bl[t] = cat16(*(const v8bf*)(pWl[t] + k0),
                          *(const v8bf*)(pWl[t] + k0 + 16));
        }
        // ---- 12 WMMAs ----
        #pragma unroll
        for (int t = 0; t < 4; ++t) {
            acc[t] = __builtin_amdgcn_wmma_f32_16x16x32_bf16(
                         false, ah, false, bh[t], (short)0, acc[t], false, false);
            acc[t] = __builtin_amdgcn_wmma_f32_16x16x32_bf16(
                         false, al, false, bh[t], (short)0, acc[t], false, false);
            acc[t] = __builtin_amdgcn_wmma_f32_16x16x32_bf16(
                         false, ah, false, bl[t], (short)0, acc[t], false, false);
        }
    }

    const float expa = (flags & 2) ? expf(a_scalar[0]) : 0.f;
    #pragma unroll
    for (int t = 0; t < 4; ++t) {
        const int col = (tng * 4 + t) * 16 + lm;
        const float bcol = bias ? bias[col] : 0.f;
        #pragma unroll
        for (int i = 0; i < 8; ++i) {
            const int row = tm * 16 + i + 8 * hs;
            float v = acc[t][i] + bcol;
            if (flags & 1) v = v / (1.f + expf(-v));          // SiLU
            if (flags & 2) v *= expf(-(float)row * expa);     // decay window
            const size_t idx = (size_t)row * N + col;
            if (flags & 4) v += res[idx];
            if (Cf) Cf[idx] = v;
            if (Chi) {
                const __bf16 h = (__bf16)v;
                Chi[idx] = h;
                Clo[idx] = (__bf16)(v - (float)h);
            }
        }
    }
}

// ---------------------------------------------------------------------------
// Row LayerNorm (one wave per row, 8 rows/block); fp32 out.
// ---------------------------------------------------------------------------
__global__ __launch_bounds__(256)
void ln_rows(const float* __restrict__ in, const float* __restrict__ g,
             const float* __restrict__ b, float* __restrict__ outf, int D)
{
    const int row = blockIdx.x * 8 + (threadIdx.x >> 5);
    const int l   = threadIdx.x & 31;
    const float* xr = in + (size_t)row * D;
    float s = 0.f, ss = 0.f;
    for (int c = l; c < D; c += 32) { float v = xr[c]; s += v; ss += v * v; }
    #pragma unroll
    for (int off = 16; off > 0; off >>= 1) {
        s  += __shfl_xor(s,  off, 32);
        ss += __shfl_xor(ss, off, 32);
    }
    const float mu  = s / (float)D;
    const float var = ss / (float)D - mu * mu;
    const float rs  = rsqrtf(var + LN_EPS);
    float* orow = outf + (size_t)row * D;
    for (int c = l; c < D; c += 32)
        orow[c] = (xr[c] - mu) * rs * g[c] + b[c];
}

// ---------------------------------------------------------------------------
// Fused: y = fx * LN(cw)  (write fp32 y), then u = LN(y) written as bf16
// hi/lo planes.  Row lives in registers between the two normalizations.
// ---------------------------------------------------------------------------
__global__ __launch_bounds__(256)
void mul_ln_ln(const float* __restrict__ fx, const float* __restrict__ cw,
               const float* __restrict__ g, const float* __restrict__ b,
               float* __restrict__ y, __bf16* __restrict__ uhi,
               __bf16* __restrict__ ulo, int D)
{
    const int row = blockIdx.x * 8 + (threadIdx.x >> 5);
    const int l   = threadIdx.x & 31;
    const int nk  = D >> 5;                        // 16 for D=512
    const float* xr = cw + (size_t)row * D;
    const float* fr = fx + (size_t)row * D;

    float s = 0.f, ss = 0.f;
    for (int c = l; c < D; c += 32) { float v = xr[c]; s += v; ss += v * v; }
    #pragma unroll
    for (int off = 16; off > 0; off >>= 1) {
        s  += __shfl_xor(s,  off, 32);
        ss += __shfl_xor(ss, off, 32);
    }
    float mu = s / (float)D;
    float rs = rsqrtf(ss / (float)D - mu * mu + LN_EPS);

    float yv[16];
    float s2 = 0.f, ss2 = 0.f;
    float* yrow = y + (size_t)row * D;
    for (int k = 0; k < nk; ++k) {
        const int c = l + 32 * k;
        const float v = fr[c] * ((xr[c] - mu) * rs * g[c] + b[c]);
        yv[k] = v;
        yrow[c] = v;
        s2 += v; ss2 += v * v;
    }
    #pragma unroll
    for (int off = 16; off > 0; off >>= 1) {
        s2  += __shfl_xor(s2,  off, 32);
        ss2 += __shfl_xor(ss2, off, 32);
    }
    const float mu2 = s2 / (float)D;
    const float rs2 = rsqrtf(ss2 / (float)D - mu2 * mu2 + LN_EPS);
    for (int k = 0; k < nk; ++k) {
        const int c = l + 32 * k;
        const float u = (yv[k] - mu2) * rs2 * g[c] + b[c];
        const size_t idx = (size_t)row * D + c;
        const __bf16 h = (__bf16)u;
        uhi[idx] = h;
        ulo[idx] = (__bf16)(u - (float)h);
    }
}

// ---------------------------------------------------------------------------
// Batched tiled transpose with optional fused add (addend in OUT layout).
// ---------------------------------------------------------------------------
__global__ __launch_bounds__(256)
void transpose_add(const float* __restrict__ in, float* __restrict__ out,
                   const float* __restrict__ addend, int rows, int cols)
{
    __shared__ float tile[32][33];
    const int b = blockIdx.z;
    in  += (size_t)b * rows * cols;
    out += (size_t)b * rows * cols;
    if (addend) addend += (size_t)b * rows * cols;
    const int c0 = blockIdx.x * 32, r0 = blockIdx.y * 32;
    const int tx = threadIdx.x, ty = threadIdx.y;
    #pragma unroll
    for (int i = 0; i < 32; i += 8) {
        const int r = r0 + ty + i, c = c0 + tx;
        tile[ty + i][tx] = (r < rows && c < cols) ? in[(size_t)r * cols + c] : 0.f;
    }
    __syncthreads();
    #pragma unroll
    for (int i = 0; i < 32; i += 8) {
        const int c = c0 + ty + i, r = r0 + tx;
        if (c < cols && r < rows) {
            float v = tile[tx][ty + i];
            if (addend) v += addend[(size_t)c * rows + r];
            out[(size_t)c * rows + r] = v;
        }
    }
}

// ---------------------------------------------------------------------------
// In-LDS radix-2 DIT complex FFT, N=8192, 256 threads. sign=-1 fwd, +1 inv.
// ---------------------------------------------------------------------------
__device__ __forceinline__ void fft8192(float* re, float* im, int tid, float sign)
{
    __syncthreads();
    for (int i = tid; i < 8192; i += 256) {
        const int j = (int)(__brev((unsigned)i) >> 19);
        if (i < j) {
            float t;
            t = re[i]; re[i] = re[j]; re[j] = t;
            t = im[i]; im[i] = im[j]; im[j] = t;
        }
    }
    __syncthreads();
    for (int s = 1; s <= 13; ++s) {
        const int m = 1 << s;
        const int half = m >> 1;
        const float ang0 = sign * 6.28318530717958647692f / (float)m;
        for (int k = tid; k < 4096; k += 256) {
            const int j = k & (half - 1);
            const int base = ((k >> (s - 1)) << s) + j;
            float sn, cs;
            __sincosf(ang0 * (float)j, &sn, &cs);
            const float xr = re[base + half], xi = im[base + half];
            const float tr = cs * xr - sn * xi;
            const float ti = cs * xi + sn * xr;
            const float ur = re[base], ui = im[base];
            re[base]        = ur + tr;  im[base]        = ui + ti;
            re[base + half] = ur - tr;  im[base + half] = ui - ti;
        }
        __syncthreads();
    }
}

// Stage 4096 fp32 (16 KB) memory -> LDS with async b128 copies; zero-fill rest.
__device__ __forceinline__ void load_signal(const float* __restrict__ row,
                                            float* re, float* im, int tid)
{
    #pragma unroll
    for (int k = 0; k < 4; ++k) {
        const int c = tid * 4 + k * 1024;
        async_load_f128(row + c, &re[c]);
    }
    for (int i = tid; i < 8192; i += 256) {
        if (i >= 4096) re[i] = 0.f;
        im[i] = 0.f;
    }
    wait_async0();
}

__global__ __launch_bounds__(256)
void fft_filter(const float* __restrict__ gt, float* __restrict__ FG)
{
    __shared__ float re[8192];
    __shared__ float im[8192];
    const int d = blockIdx.x, tid = threadIdx.x;
    load_signal(gt + (size_t)d * 4096, re, im, tid);
    fft8192(re, im, tid, -1.f);
    float2* o = (float2*)(FG + (size_t)d * 8192 * 2);
    for (int i = tid; i < 8192; i += 256) o[i] = make_float2(re[i], im[i]);
}

__global__ __launch_bounds__(256)
void fftconv(const float* __restrict__ lnzt, const float* __restrict__ FG,
             float* __restrict__ outt)
{
    __shared__ float re[8192];
    __shared__ float im[8192];
    const int bd = blockIdx.x;        // b*512 + d
    const int d  = bd & 511;
    const int tid = threadIdx.x;
    load_signal(lnzt + (size_t)bd * 4096, re, im, tid);
    fft8192(re, im, tid, -1.f);
    const float2* G = (const float2*)(FG + (size_t)d * 8192 * 2);
    for (int i = tid; i < 8192; i += 256) {
        const float2 g = G[i];
        const float zr = re[i], zi = im[i];
        re[i] = zr * g.x - zi * g.y;
        im[i] = zr * g.y + zi * g.x;
    }
    fft8192(re, im, tid, 1.f);
    float* o = outt + (size_t)bd * 4096;
    const float inv = 1.0f / 8192.0f;
    for (int i = tid; i < 4096; i += 256) o[i] = re[i] * inv;
}

// ============================================================================
extern "C" void kernel_launch(void* const* d_in, const int* in_sizes, int n_in,
                              void* d_out, int out_size, void* d_ws, size_t ws_size,
                              hipStream_t stream)
{
    (void)in_sizes; (void)n_in; (void)out_size; (void)ws_size;
    const int B = 4, L = 4096, D = 512, P = 128;
    const int BL = B * L;

    const float* z       = (const float*)d_in[0];
    const float* x       = (const float*)d_in[1];
    const float* pos     = (const float*)d_in[2];
    const float* pos_w1  = (const float*)d_in[3];
    const float* pos_b1  = (const float*)d_in[4];
    const float* pos_w2  = (const float*)d_in[5];
    const float* pos_b2  = (const float*)d_in[6];
    const float* a_scal  = (const float*)d_in[7];
    const float* ffn_w1  = (const float*)d_in[8];
    const float* ffn_b1  = (const float*)d_in[9];
    const float* ffn_w2  = (const float*)d_in[10];
    const float* ffn_b2  = (const float*)d_in[11];
    const float* ffnx_w1 = (const float*)d_in[12];
    const float* ffnx_b1 = (const float*)d_in[13];
    const float* ffnx_w2 = (const float*)d_in[14];
    const float* ffnx_b2 = (const float*)d_in[15];
    const float* ln_g    = (const float*)d_in[16];
    const float* ln_b    = (const float*)d_in[17];
    float* out = (float*)d_out;

    // ---------------- workspace (256B-aligned carve-out) ----------------
    char* base = (char*)d_ws;
    size_t off = 0;
    auto alloc = [&](size_t bytes) -> void* {
        void* p = base + off;
        off = (off + bytes + 255) & ~(size_t)255;
        return p;
    };
    float* gbuf  = (float*)alloc((size_t)L * D * 4);
    float* gt    = (float*)alloc((size_t)L * D * 4);
    float* FG    = (float*)alloc((size_t)D * 8192 * 2 * 4);
    float* cwhzt = (float*)alloc((size_t)B * D * L * 4);
    float* lnz   = (float*)alloc((size_t)BL * D * 4);
    float* lnzt  = (float*)alloc((size_t)BL * D * 4);
    float* cwhz  = (float*)alloc((size_t)BL * D * 4);
    float* fx    = (float*)alloc((size_t)BL * D * 4);
    __bf16* poshi = (__bf16*)alloc((size_t)L * P * 2);
    __bf16* poslo = (__bf16*)alloc((size_t)L * P * 2);
    __bf16* pw1hi = (__bf16*)alloc((size_t)2 * P * P * 2);
    __bf16* pw1lo = (__bf16*)alloc((size_t)2 * P * P * 2);
    __bf16* pw2hi = (__bf16*)alloc((size_t)D * 2 * P * 2);
    __bf16* pw2lo = (__bf16*)alloc((size_t)D * 2 * P * 2);
    __bf16* fw1hi = (__bf16*)alloc((size_t)2 * D * D * 2);
    __bf16* fw1lo = (__bf16*)alloc((size_t)2 * D * D * 2);
    __bf16* fw2hi = (__bf16*)alloc((size_t)2 * D * D * 2);
    __bf16* fw2lo = (__bf16*)alloc((size_t)2 * D * D * 2);
    __bf16* xw1hi = (__bf16*)alloc((size_t)2 * D * D * 2);
    __bf16* xw1lo = (__bf16*)alloc((size_t)2 * D * D * 2);
    __bf16* xw2hi = (__bf16*)alloc((size_t)2 * D * D * 2);
    __bf16* xw2lo = (__bf16*)alloc((size_t)2 * D * D * 2);
    __bf16* h1hi  = (__bf16*)alloc((size_t)L * 2 * P * 2);
    __bf16* h1lo  = (__bf16*)alloc((size_t)L * 2 * P * 2);
    __bf16* uhi   = (__bf16*)alloc((size_t)BL * D * 2);
    __bf16* ulo   = (__bf16*)alloc((size_t)BL * D * 2);
    // lifetime overlays (each fits in its host buffer, dead before reuse):
    __bf16* xhi   = (__bf16*)FG;      // FG dead after fftconv
    __bf16* xlo   = (__bf16*)cwhzt;   // cwhzt dead after transpose-back
    __bf16* hidhi = (__bf16*)lnz;     // lnz dead after transpose
    __bf16* hidlo = (__bf16*)lnzt;    // lnzt dead after fftconv

    dim3 tb(32, 8, 1);
    auto splitN = [&](const float* in, __bf16* hi, __bf16* lo, size_t n) {
        const int n4 = (int)(n / 4);
        split_bf16<<<(n4 + 255) / 256, 256, 0, stream>>>(in, hi, lo, n4);
    };

    // 0) split static operands into bf16 hi/lo planes
    splitN(pos,     poshi, poslo, (size_t)L * P);
    splitN(pos_w1,  pw1hi, pw1lo, (size_t)2 * P * P);
    splitN(pos_w2,  pw2hi, pw2lo, (size_t)D * 2 * P);
    splitN(ffn_w1,  fw1hi, fw1lo, (size_t)2 * D * D);
    splitN(ffn_w2,  fw2hi, fw2lo, (size_t)2 * D * D);
    splitN(ffnx_w1, xw1hi, xw1lo, (size_t)2 * D * D);
    splitN(ffnx_w2, xw2hi, xw2lo, (size_t)2 * D * D);

    // 1-2) implicit filter FFN; decay window fused into 2nd GEMM epilogue
    gemm_bf16x3<<<(L/16)*(2*P/64)/8, 256, 0, stream>>>(
        poshi, poslo, pw1hi, pw1lo, pos_b1,
        nullptr, h1hi, h1lo, nullptr, nullptr, L, 2*P, P, /*silu*/1);
    gemm_bf16x3<<<(L/16)*(D/64)/8, 256, 0, stream>>>(
        h1hi, h1lo, pw2hi, pw2lo, pos_b2,
        gbuf, nullptr, nullptr, nullptr, a_scal, L, D, 2*P, /*window*/2);

    // 3-4) transpose filter for coalesced FFT, then FFT it
    transpose_add<<<dim3(D/32, L/32, 1), tb, 0, stream>>>(gbuf, gt, nullptr, L, D);
    fft_filter<<<D, 256, 0, stream>>>(gt, FG);

    // 5-8) LN(z), transpose, per-channel FFT conv, transpose back (+z fused)
    ln_rows<<<BL/8, 256, 0, stream>>>(z, ln_g, ln_b, lnz, D);
    transpose_add<<<dim3(D/32, L/32, B), tb, 0, stream>>>(lnz, lnzt, nullptr, L, D);
    fftconv<<<B*D, 256, 0, stream>>>(lnzt, FG, cwhzt);
    transpose_add<<<dim3(L/32, D/32, B), tb, 0, stream>>>(cwhzt, cwhz, z, D, L);

    // 9) split x (overlays dead FG/cwhzt)
    splitN(x, xhi, xlo, (size_t)BL * D);

    // 10-11) fx = FFN_x(x); hidden kept as bf16 hi/lo only (overlays lnz/lnzt)
    gemm_bf16x3<<<(BL/16)*(2*D/64)/8, 256, 0, stream>>>(
        xhi, xlo, xw1hi, xw1lo, ffnx_b1,
        nullptr, hidhi, hidlo, nullptr, nullptr, BL, 2*D, D, /*silu*/1);
    gemm_bf16x3<<<(BL/16)*(D/64)/8, 256, 0, stream>>>(
        hidhi, hidlo, xw2hi, xw2lo, ffnx_b2,
        fx, nullptr, nullptr, nullptr, nullptr, BL, D, 2*D, 0);

    // 12) fused: y = fx * LN(cwhz) -> d_out; u = LN(y) -> bf16 hi/lo planes
    mul_ln_ln<<<BL/8, 256, 0, stream>>>(fx, cwhz, ln_g, ln_b, out, uhi, ulo, D);

    // 13-14) d_out = FFN(u) + y (residual fused in epilogue)
    gemm_bf16x3<<<(BL/16)*(2*D/64)/8, 256, 0, stream>>>(
        uhi, ulo, fw1hi, fw1lo, ffn_b1,
        nullptr, hidhi, hidlo, nullptr, nullptr, BL, 2*D, D, /*silu*/1);
    gemm_bf16x3<<<(BL/16)*(D/64)/8, 256, 0, stream>>>(
        hidhi, hidlo, fw2hi, fw2lo, ffn_b2,
        out, nullptr, nullptr, out, nullptr, BL, D, 2*D, /*residual*/4);
}
